// NCELoss_22539988370160
// MI455X (gfx1250) — compile-verified, matrix operand order checked by reading, not run on hardware
//
#include <hip/hip_runtime.h>
#include <math.h>

// ---------------------------------------------------------------------------
// NCE loss: f0,f1 [B,N,D] fp32; corrs [B,M,2]; loss = mean_{b,j}(LSE_i - diag)
//   B=8, N=20000, D=128, M=1024, T=0.07
// Pipeline:
//   1) gather + L2-normalize only the M selected rows; split fp32 -> f16 hi/lo
//   2) fused WMMA GEMM (compensated f16 ~= fp32 accuracy) + online column-LSE
//      - 4 waves/block, each sweeps 16 row tiles
//      - 3-stage ring buffer, prefetch distance 2 K-chunks (8 loads in flight)
//      - per-column (m,s) merged across waves via LDS
//   3) deterministic tree reduce of per-block partials
// ---------------------------------------------------------------------------

typedef __attribute__((ext_vector_type(16))) _Float16 v16h;
typedef __attribute__((ext_vector_type(8)))  _Float16 v8h;
typedef __attribute__((ext_vector_type(4)))  _Float16 v4h;
typedef __attribute__((ext_vector_type(8)))  float    v8f;

constexpr int   B_  = 8;
constexpr int   N_  = 20000;
constexpr int   D_  = 128;
constexpr int   M_  = 1024;
constexpr int   COL_TILES = M_ / 16;          // 64 tiles of 16 per batch
constexpr int   WAVES     = 4;                // waves per GEMM block
constexpr int   RT_PER_W  = COL_TILES / WAVES;// 16 row tiles per wave
constexpr int   STEPS     = RT_PER_W * 4;     // (row tile, K-chunk) steps
constexpr float INV_T = 1.0f / 0.07f;

// -------------------------------------------------------------------- WMMA --
__device__ __forceinline__ v8f wmma_f16(v16h a, v16h b, v8f c) {
    // 8 args: (neg_a, A, neg_b, B, c_mod, C, reuse_a, reuse_b)
    return __builtin_amdgcn_wmma_f32_16x16x32_f16(false, a, false, b,
                                                  (short)0, c, false, false);
}

// Load one 16x16x32-f16 A/B fragment for this lane from a K-major row.
// Lane l holds row/col (l&15); K slices depend on s = l>>4:
//   halves [kc*32 + 8*s .. +7] and [kc*32 + 16 + 8*s .. +7]  -> two b128 loads
__device__ __forceinline__ v16h load_frag(const _Float16* __restrict__ rowbase,
                                          int kc, int s) {
    const v8h* p0 = reinterpret_cast<const v8h*>(rowbase + kc * 32 + 8 * s);
    const v8h* p1 = reinterpret_cast<const v8h*>(rowbase + kc * 32 + 16 + 8 * s);
    v8h a = *p0;
    v8h b = *p1;
    return __builtin_shufflevector(a, b, 0, 1, 2, 3, 4, 5, 6, 7,
                                         8, 9, 10, 11, 12, 13, 14, 15);
}

struct KFrag { v16h h; v16h l; };        // hi/lo fragments for one K-chunk

// compensated product: acc += (ah+al) * (bh+bl), dropping al*bl (~2^-22 rel)
__device__ __forceinline__ v8f dot3(const KFrag& a, v16h bh, v16h bl, v8f acc) {
    acc = wmma_f16(a.l, bh, acc);
    acc = wmma_f16(a.h, bl, acc);
    acc = wmma_f16(a.h, bh, acc);
    return acc;
}

// ------------------------------------------------- 1) gather/normalize/split
__device__ __forceinline__ void norm_split_row(const float* __restrict__ src,
                                               _Float16* __restrict__ dh,
                                               _Float16* __restrict__ dl,
                                               int lane) {
    float4 v = reinterpret_cast<const float4*>(src)[lane];
    float ss = v.x * v.x + v.y * v.y + v.z * v.z + v.w * v.w;
#pragma unroll
    for (int o = 16; o > 0; o >>= 1) ss += __shfl_xor(ss, o, 32);
    float inv = 1.0f / fmaxf(sqrtf(ss), 1e-12f);
    float y[4] = {v.x * inv, v.y * inv, v.z * inv, v.w * inv};
    v4h h, l;
#pragma unroll
    for (int i = 0; i < 4; ++i) {
        _Float16 hi = (_Float16)y[i];
        h[i] = hi;
        l[i] = (_Float16)(y[i] - (float)hi);
    }
    reinterpret_cast<v4h*>(dh)[lane] = h;
    reinterpret_cast<v4h*>(dl)[lane] = l;
}

__global__ void gather_norm_split(const float* __restrict__ feats0,
                                  const float* __restrict__ feats1,
                                  const int*   __restrict__ corrs,
                                  _Float16* __restrict__ f0h,
                                  _Float16* __restrict__ f0l,
                                  _Float16* __restrict__ f1h,
                                  _Float16* __restrict__ f1l) {
    const int wave = threadIdx.x >> 5;
    const int lane = threadIdx.x & 31;
    const int idx  = blockIdx.x * 8 + wave;        // 0 .. B*M-1
    if (idx >= B_ * M_) return;
    const int b  = idx / M_;
    const int i0 = corrs[idx * 2 + 0];
    const int i1 = corrs[idx * 2 + 1];

    const float* s0 = feats0 + ((size_t)b * N_ + (size_t)i0) * D_;
    const float* s1 = feats1 + ((size_t)b * N_ + (size_t)i1) * D_;
    const size_t roff = (size_t)idx * D_;
    norm_split_row(s0, f0h + roff, f0l + roff, lane);
    norm_split_row(s1, f1h + roff, f1l + roff, lane);
}

// --------------------------------------- 2) fused WMMA GEMM + online col-LSE
// grid = B * COL_TILES blocks, 128 threads (4 waves). Column tile = f0 side.
__global__ void __launch_bounds__(WAVES * 32)
nce_gemm_lse(const _Float16* __restrict__ f0h, const _Float16* __restrict__ f0l,
             const _Float16* __restrict__ f1h, const _Float16* __restrict__ f1l,
             float* __restrict__ partials) {
    const int ct   = blockIdx.x % COL_TILES;
    const int b    = blockIdx.x / COL_TILES;
    const int wq   = threadIdx.x >> 5;   // wave index 0..3
    const int lane = threadIdx.x & 31;
    const int c    = lane & 15;          // column / row within tile
    const int s    = lane >> 4;          // K-slice selector

    const size_t base = (size_t)b * M_ * D_;

    // Preload B-side (f0, columns) fragments: 4 K-chunks, hi+lo (64 VGPRs)
    const _Float16* browh = f0h + base + (size_t)(ct * 16 + c) * D_;
    const _Float16* browl = f0l + base + (size_t)(ct * 16 + c) * D_;
    v16h Bh[4], Bl[4];
#pragma unroll
    for (int kc = 0; kc < 4; ++kc) {
        Bh[kc] = load_frag(browh, kc, s);
        Bl[kc] = load_frag(browl, kc, s);
    }

    // A-side row range for this wave
    const int rt0 = wq * RT_PER_W;
    const _Float16* ah = f1h + base + (size_t)(rt0 * 16 + c) * D_;
    const _Float16* al = f1l + base + (size_t)(rt0 * 16 + c) * D_;
    constexpr int ROWSTEP = 16 * D_;     // halves between row tiles

    float mrun = -INFINITY;   // running column max (per-lane partial view)
    float srun = 0.0f;        // running sum of exp(x - mrun)
    float diag = 0.0f;        // diagonal logit owned by this lane (if any)
    const bool owner = (s == 0) ? (c < 8) : (c >= 8);
    const int  want  = (s == 0) ? c : (c - 8);

    // 3-stage ring buffer over flat (row-tile, K-chunk) steps; distance 2.
    KFrag ring[3];
#define LOAD_STEP(t)                                                         \
    do {                                                                     \
        const _Float16* rh_ = ah + (size_t)((t) >> 2) * ROWSTEP;             \
        const _Float16* rl_ = al + (size_t)((t) >> 2) * ROWSTEP;             \
        ring[(t) % 3].h = load_frag(rh_, (t) & 3, s);                        \
        ring[(t) % 3].l = load_frag(rl_, (t) & 3, s);                        \
    } while (0)

    LOAD_STEP(0);
    LOAD_STEP(1);
    v8f acc = {};
#pragma unroll
    for (int t = 0; t < STEPS; ++t) {
        if (t + 2 < STEPS) LOAD_STEP(t + 2);         // keep 8 loads in flight
        acc = dot3(ring[t % 3], Bh[t & 3], Bl[t & 3], acc);
        if ((t & 3) == 3) {
            // finished row tile rt: scale by 1/T, online softmax update
            const int rt = rt0 + (t >> 2);
            float x[8];
            float tmax = -INFINITY;
#pragma unroll
            for (int j = 0; j < 8; ++j) {
                x[j] = acc[j] * INV_T;
                tmax = fmaxf(tmax, x[j]);
            }
            const float nm = fmaxf(mrun, tmax);
            float se = 0.0f;
#pragma unroll
            for (int j = 0; j < 8; ++j) se += __expf(x[j] - nm);
            srun = srun * __expf(mrun - nm) + se;
            mrun = nm;

            if (rt == ct) {            // diagonal tile: grab logit[j,j]
#pragma unroll
                for (int j = 0; j < 8; ++j)
                    if (owner && want == j) diag = x[j];
            }
            acc = v8f{};
        }
    }
#undef LOAD_STEP

    // Merge the two lanes covering each column (lane c and lane c+16)
    const float m2 = __shfl_xor(mrun, 16, 32);
    const float s2 = __shfl_xor(srun, 16, 32);
    const float d2 = __shfl_xor(diag, 16, 32);
    const float mm = fmaxf(mrun, m2);
    const float S  = srun * __expf(mrun - mm) + s2 * __expf(m2 - mm);

    // Cross-wave merge via LDS
    __shared__ float lm[WAVES][16];
    __shared__ float ls[WAVES][16];
    __shared__ float ld[16];
    if (lane < 16) {
        lm[wq][c] = mm;
        ls[wq][c] = S;
        if (wq == (ct >> 4))                       // wave that owns diag tile
            ld[c] = (c < 8) ? diag : d2;
    }
    __syncthreads();

    if (wq == 0) {
        float contrib = 0.0f;
        if (lane < 16) {
            float M = lm[0][c];
#pragma unroll
            for (int w = 1; w < WAVES; ++w) M = fmaxf(M, lm[w][c]);
            float Ssum = 0.0f;
#pragma unroll
            for (int w = 0; w < WAVES; ++w)
                Ssum += ls[w][c] * __expf(lm[w][c] - M);
            contrib = (M + __logf(Ssum)) - ld[c];
        }
#pragma unroll
        for (int o = 16; o > 0; o >>= 1) contrib += __shfl_xor(contrib, o, 32);
        if (lane == 0)
            partials[blockIdx.x] = contrib * (1.0f / (float)(B_ * M_));
    }
}

// ----------------------------------------------------- 3) deterministic sum
__global__ void reduce_partials(const float* __restrict__ partials,
                                float* __restrict__ out) {
    __shared__ float sm[B_ * COL_TILES];
    const int t = threadIdx.x;
    sm[t] = partials[t];
    __syncthreads();
#pragma unroll
    for (int st = (B_ * COL_TILES) / 2; st > 0; st >>= 1) {
        if (t < st) sm[t] += sm[t + st];
        __syncthreads();
    }
    if (t == 0) out[0] = sm[0];
}

// ---------------------------------------------------------------------------
extern "C" void kernel_launch(void* const* d_in, const int* in_sizes, int n_in,
                              void* d_out, int out_size, void* d_ws, size_t ws_size,
                              hipStream_t stream) {
    const float* feats0 = (const float*)d_in[0];
    const float* feats1 = (const float*)d_in[1];
    const int*   corrs  = (const int*)d_in[2];

    const size_t sz = (size_t)B_ * M_ * D_;            // halves per buffer (2 MB)
    _Float16* f0h = (_Float16*)d_ws;
    _Float16* f0l = f0h + sz;
    _Float16* f1h = f0l + sz;
    _Float16* f1l = f1h + sz;
    float* partials = (float*)(f1l + sz);              // 512 floats

    gather_norm_split<<<(B_ * M_) / 8, 256, 0, stream>>>(
        feats0, feats1, corrs, f0h, f0l, f1h, f1l);

    nce_gemm_lse<<<B_ * COL_TILES, WAVES * 32, 0, stream>>>(
        f0h, f0l, f1h, f1l, partials);

    reduce_partials<<<1, B_ * COL_TILES, 0, stream>>>(partials, (float*)d_out);
}
